// MSA2Pair_30193620091046
// MI455X (gfx1250) — matrix operand order repaired and checked by reading, not compiled, not run on hardware
//
#include <hip/hip_runtime.h>
#include <hip/hip_bf16.h>

typedef __attribute__((ext_vector_type(16))) __bf16 v16bf;
typedef __attribute__((ext_vector_type(8)))  float  v8f;

union BF16Frag { v16bf v; unsigned short u[16]; };

#define LN_EPS 1e-5f
#define IN_EPS 1e-6f

// Native f32 -> bf16 (RNE) conversion; lowers to gfx1250 hardware convert.
__device__ __forceinline__ unsigned short f2bf(float f) {
  __bf16 h = (__bf16)f;
  return __builtin_bit_cast(unsigned short, h);
}
__device__ __forceinline__ float wave_sum(float v) {
#pragma unroll
  for (int m = 16; m > 0; m >>= 1) v += __shfl_xor(v, m, 32);
  return v;
}
__device__ __forceinline__ float eluf(float v) { return v > 0.f ? v : expm1f(v); }

// Load one 32-byte lane-fragment with two 128-bit loads.
#define LOAD_FRAG(frag, ptr)                                      \
  do {                                                            \
    const uint4* _p = (const uint4*)(ptr);                        \
    ((uint4*)(frag).u)[0] = _p[0];                                \
    ((uint4*)(frag).u)[1] = _p[1];                                \
  } while (0)

// ---------------------------------------------------------------------------
// Weight preconversion into WMMA fragment-major order (bf16).
// A-operand fragment layout (16x32, 16-bit):
//   lane = (M&15) + 16*(Klocal>>4);      elem = Klocal & 15
// B-operand fragment layout (32x16, 16-bit):
//   lane = 16*((Klocal>>3)&1) + (N&15); elem = (Klocal&7) + 8*(Klocal>>4)
// ---------------------------------------------------------------------------
// w2[o][k] (128x1024) -> B-frags dst[ks(32)][nt(8)][lane(32)][elem(16)]
__global__ void k_prep_w2frag(const float* __restrict__ w2,
                              unsigned short* __restrict__ dst) {
  int i = blockIdx.x * 256 + threadIdx.x;          // 131072
  int e = i & 15, ln = (i >> 4) & 31, nt = (i >> 9) & 7, ks = i >> 12;
  int o = nt * 16 + (ln & 15);
  int k = ks * 32 + ((ln >> 4) << 3) + (e < 8 ? e : e + 8);
  dst[i] = f2bf(w2[o * 1024 + k]);
}
// conv_in_w[m][k] (128x384) -> A-frags dst[ks(12)][mt(8)][lane][elem]
__global__ void k_prep_w1x1frag(const float* __restrict__ w,
                                unsigned short* __restrict__ dst) {
  int i = blockIdx.x * 256 + threadIdx.x;          // 49152
  int e = i & 15, ln = (i >> 4) & 31, mt = (i >> 9) & 7, ks = i >> 12;
  int kk = ((ln >> 4) << 4) + e;
  int m = mt * 16 + (ln & 15);
  dst[i] = f2bf(w[m * 384 + ks * 32 + kk]);
}
// rb_w[m][c][3][3] -> A-frags dst[kst(36=cc*9+tap)][mt(8)][lane][elem]
__global__ void k_prep_w3x3frag(const float* __restrict__ w,
                                unsigned short* __restrict__ dst) {
  int i = blockIdx.x * 256 + threadIdx.x;          // 147456
  int e = i & 15, ln = (i >> 4) & 31, mt = (i >> 9) & 7, kst = i >> 12;
  int cc = kst / 9, tap = kst % 9;
  int kk = ((ln >> 4) << 4) + e;
  int m = mt * 16 + (ln & 15);
  dst[i] = f2bf(w[m * 1152 + (cc * 32 + kk) * 9 + tap]);
}

// ---------------------------------------------------------------------------
// LayerNorm(msa, F=64) + proj to P=32.  One wave per (n,l) row.
// ---------------------------------------------------------------------------
__global__ __launch_bounds__(256) void k_ln1_proj(
    const float* __restrict__ msa,
    const float* __restrict__ g, const float* __restrict__ b,
    const float* __restrict__ w1, const float* __restrict__ b1,
    float* __restrict__ xdf, unsigned short* __restrict__ xdb) {
  __shared__ float rowbuf[8][64];
  int w = threadIdx.x >> 5, lane = threadIdx.x & 31;
  int row = blockIdx.x * 8 + w;                 // row = n*256 + l
  float a0 = msa[row * 64 + lane];
  float a1 = msa[row * 64 + 32 + lane];
  float mean = wave_sum(a0 + a1) * (1.0f / 64.0f);
  float d0 = a0 - mean, d1 = a1 - mean;
  float var = wave_sum(d0 * d0 + d1 * d1) * (1.0f / 64.0f);
  float rstd = rsqrtf(var + LN_EPS);
  rowbuf[w][lane]      = d0 * rstd * g[lane]      + b[lane];
  rowbuf[w][lane + 32] = d1 * rstd * g[lane + 32] + b[lane + 32];
  __syncthreads();
  float acc = b1[lane];
#pragma unroll
  for (int f = 0; f < 64; ++f) acc += rowbuf[w][f] * w1[lane * 64 + f];
  int n = row >> 8, l = row & 255;
  xdf[row * 32 + lane] = acc;
  xdb[n * 8192 + l * 32 + lane] = f2bf(acc);
}

// ---------------------------------------------------------------------------
// feat_1d = concat(mean_n LN(x_down), LN(x_down)[n=0]) -> featb [L][64] bf16
// ---------------------------------------------------------------------------
__global__ __launch_bounds__(256) void k_feat1d(
    const float* __restrict__ xdf,
    const float* __restrict__ g, const float* __restrict__ b,
    unsigned short* __restrict__ featb) {
  __shared__ float facc[8][32];
  int w = threadIdx.x >> 5, lane = threadIdx.x & 31;
  int l = blockIdx.x;
  float acc = 0.f;
  for (int it = 0; it < 24; ++it) {
    int n = w + it * 8;
    float x = xdf[(n * 256 + l) * 32 + lane];
    float mean = wave_sum(x) * (1.0f / 32.0f);
    float d = x - mean;
    float var = wave_sum(d * d) * (1.0f / 32.0f);
    float nv = d * rsqrtf(var + LN_EPS) * g[lane] + b[lane];
    acc += nv;
    if (n == 0) featb[l * 64 + 32 + lane] = f2bf(nv);
  }
  facc[w][lane] = acc;
  __syncthreads();
  if (w == 0) {
    float t = 0.f;
#pragma unroll
    for (int i = 0; i < 8; ++i) t += facc[i][lane];
    featb[l * 64 + lane] = f2bf(t * (1.0f / 192.0f));
  }
}

// ---------------------------------------------------------------------------
// Row-wise LayerNorm over 128 channels -> bf16 (pair_orig)
// ---------------------------------------------------------------------------
__global__ __launch_bounds__(256) void k_row_ln128_bf16(
    const float* __restrict__ src,
    const float* __restrict__ g, const float* __restrict__ b,
    unsigned short* __restrict__ dst) {
  int w = threadIdx.x >> 5, lane = threadIdx.x & 31;
  long s = (long)blockIdx.x * 8 + w;
  const float* r = src + s * 128;
  float v0 = r[lane], v1 = r[lane + 32], v2 = r[lane + 64], v3 = r[lane + 96];
  float mean = wave_sum(v0 + v1 + v2 + v3) * (1.0f / 128.0f);
  float d0 = v0 - mean, d1 = v1 - mean, d2 = v2 - mean, d3 = v3 - mean;
  float var = wave_sum(d0 * d0 + d1 * d1 + d2 * d2 + d3 * d3) * (1.0f / 128.0f);
  float rstd = rsqrtf(var + LN_EPS);
  unsigned short* o = dst + s * 128;
  o[lane]      = f2bf(d0 * rstd * g[lane]      + b[lane]);
  o[lane + 32] = f2bf(d1 * rstd * g[lane + 32] + b[lane + 32]);
  o[lane + 64] = f2bf(d2 * rstd * g[lane + 64] + b[lane + 64]);
  o[lane + 96] = f2bf(d3 * rstd * g[lane + 96] + b[lane + 96]);
}

// ---------------------------------------------------------------------------
// Fused: 128x128x192 outer-product WMMA GEMM -> LN over 1024 per (i,l)
//        -> x w2 (WMMA, K=1024) + b2 -> LN_new -> pn_b bf16 [L*L][128]
// Block = 4 i x 4 l pairs; wave = 2 M-tiles x 4 N-tiles.
// ---------------------------------------------------------------------------
__global__ __launch_bounds__(256) void k_pair_fused(
    const unsigned short* __restrict__ xdb,
    const float* __restrict__ g2d, const float* __restrict__ b2d,
    const unsigned short* __restrict__ w2f,   // B-frag-major weights
    const float* __restrict__ b2,
    const float* __restrict__ lnn_g, const float* __restrict__ lnn_b,
    unsigned short* __restrict__ pn_b) {
  __shared__ __align__(16) unsigned short Af[8][32][16];    // 8 KB
  __shared__ __align__(16) unsigned short Bf[8][32][16];    // 8 KB
  __shared__ __align__(16) unsigned short lnpF[32][32][16]; // 32 KB (GEMM2 A-frags)
  __shared__ __align__(16) float outr[16][128];             // 8 KB
  __shared__ float psum[16], psq[16];

  int w = threadIdx.x >> 5, lane = threadIdx.x & 31;
  int bi = blockIdx.y, bl = blockIdx.x;
  int ibase = bi * 128, lbase = bl * 128;
  int mt0 = (w & 3) * 2, nt0 = (w >> 2) * 4;

  v8f acc[2][4] = {};
  int sk = threadIdx.x >> 3;          // staging k-row 0..31
  int sc = threadIdx.x & 7;           // staging 16-col chunk == tile idx

  // ---- GEMM1 ----
  for (int ks = 0; ks < 6; ++ks) {
    __syncthreads();
    {
      unsigned short ta[16], tb[16];
      const uint4* sa = (const uint4*)(xdb + (ks * 32 + sk) * 8192 + ibase + sc * 16);
      ((uint4*)ta)[0] = sa[0]; ((uint4*)ta)[1] = sa[1];
      const uint4* sb = (const uint4*)(xdb + (ks * 32 + sk) * 8192 + lbase + sc * 16);
      ((uint4*)tb)[0] = sb[0]; ((uint4*)tb)[1] = sb[1];
      int alane = (sk >> 4) << 4;
      int aelem = sk & 15;
      int blane = (sk >> 4) << 4;
      int belem = sk & 15;
#pragma unroll
      for (int m = 0; m < 16; ++m) Af[sc][alane + m][aelem] = ta[m];
#pragma unroll
      for (int n = 0; n < 16; ++n) Bf[sc][blane + n][belem] = tb[n];
    }
    __syncthreads();
    BF16Frag a0, a1;
    LOAD_FRAG(a0, &Af[mt0][lane][0]);
    LOAD_FRAG(a1, &Af[mt0 + 1][lane][0]);
#pragma unroll
    for (int nj = 0; nj < 4; ++nj) {
      BF16Frag bf;
      LOAD_FRAG(bf, &Bf[nt0 + nj][lane][0]);
      acc[0][nj] = __builtin_amdgcn_wmma_f32_16x16x32_bf16(
          false, a0.v, false, bf.v, (short)0, acc[0][nj], false, false);
      acc[1][nj] = __builtin_amdgcn_wmma_f32_16x16x32_bf16(
          false, a1.v, false, bf.v, (short)0, acc[1][nj], false, false);
    }
  }
  __syncthreads();

  // ---- LN statistics: each wave covers i-local (w&3) and 2 l-locals ----
#pragma unroll
  for (int lg = 0; lg < 2; ++lg) {
    float s = 0.f, q = 0.f;
#pragma unroll
    for (int mi = 0; mi < 2; ++mi)
#pragma unroll
      for (int t = 0; t < 2; ++t) {
        int nj = lg * 2 + t;
#pragma unroll
        for (int r = 0; r < 8; ++r) { float v = acc[mi][nj][r]; s += v; q += v * v; }
      }
    s = wave_sum(s); q = wave_sum(q);
    if (lane == 0) {
      int pair = (w & 3) * 4 + (w >> 2) * 2 + lg;
      psum[pair] = s; psq[pair] = q;
    }
  }
  __syncthreads();

  // ---- normalize, write straight into GEMM2 A-fragment layout ----
  const float invN = 1.0f / 192.0f;
#pragma unroll
  for (int mi = 0; mi < 2; ++mi) {
#pragma unroll
    for (int nj = 0; nj < 4; ++nj) {
      int N = (nt0 + nj) * 16 + (lane & 15);
      int pair = (w & 3) * 4 + (N >> 5);
      float mean = psum[pair] * (invN / 1024.0f);
      float ex2  = psq[pair]  * (invN * invN / 1024.0f);
      float rstd = rsqrtf(ex2 - mean * mean + LN_EPS);
#pragma unroll
      for (int r = 0; r < 8; ++r) {
        int M = (mt0 + mi) * 16 + r + ((lane >> 4) << 3);
        int pq = (M & 31) * 32 + (N & 31);
        float v = acc[mi][nj][r] * invN;
        float nv = (v - mean) * rstd * g2d[pq] + b2d[pq];
        lnpF[pq >> 5][pair + (((pq >> 3) & 1) << 4)]
            [(pq & 7) + (((pq >> 4) & 1) << 3)] = f2bf(nv);
      }
    }
  }
  __syncthreads();

  // ---- GEMM2: [16 x 1024] x [1024 x 128]; wave w = o-tile w ----
  v8f acc2 = {};
  for (int ks = 0; ks < 32; ++ks) {
    BF16Frag af, bf;
    LOAD_FRAG(af, &lnpF[ks][lane][0]);
    LOAD_FRAG(bf, w2f + (((ks * 8) + w) * 32 + lane) * 16);
    acc2 = __builtin_amdgcn_wmma_f32_16x16x32_bf16(
        false, af.v, false, bf.v, (short)0, acc2, false, false);
  }
  {
    int o = w * 16 + (lane & 15);
    float bias = b2[o];
#pragma unroll
    for (int r = 0; r < 8; ++r)
      outr[r + ((lane >> 4) << 3)][o] = acc2[r] + bias;
  }
  __syncthreads();

  // ---- LN over O=128 per pair row, emit bf16 ----
#pragma unroll
  for (int t = 0; t < 2; ++t) {
    int srow = w * 2 + t;
    float v0 = outr[srow][lane];
    float v1 = outr[srow][lane + 32];
    float v2 = outr[srow][lane + 64];
    float v3 = outr[srow][lane + 96];
    float mean = wave_sum(v0 + v1 + v2 + v3) * (1.0f / 128.0f);
    float d0 = v0 - mean, d1 = v1 - mean, d2 = v2 - mean, d3 = v3 - mean;
    float var = wave_sum(d0 * d0 + d1 * d1 + d2 * d2 + d3 * d3) * (1.0f / 128.0f);
    float rstd = rsqrtf(var + LN_EPS);
    int i = bi * 4 + (srow >> 2), l = bl * 4 + (srow & 3);
    unsigned short* o = pn_b + (long)(i * 256 + l) * 128;
    o[lane]      = f2bf(d0 * rstd * lnn_g[lane]      + lnn_b[lane]);
    o[lane + 32] = f2bf(d1 * rstd * lnn_g[lane + 32] + lnn_b[lane + 32]);
    o[lane + 64] = f2bf(d2 * rstd * lnn_g[lane + 64] + lnn_b[lane + 64]);
    o[lane + 96] = f2bf(d3 * rstd * lnn_g[lane + 96] + lnn_b[lane + 96]);
  }
}

// ---------------------------------------------------------------------------
// conv_in 1x1 as WMMA GEMM: M=128, K=384, N=64 spatial.  Concat assembled
// per-k-slab directly into B-fragment layout in LDS.
// ---------------------------------------------------------------------------
__global__ __launch_bounds__(256) void k_conv1x1(
    const unsigned short* __restrict__ w1f,    // A-frag-major [12][8][32][16]
    const unsigned short* __restrict__ po_b,   // [65536][128]
    const unsigned short* __restrict__ pn_b,   // [65536][128]
    const unsigned short* __restrict__ featb,  // [256][64]
    float* __restrict__ x0) {                  // [128][65536]
  __shared__ __align__(16) unsigned short Xf[4][32][16];
  int w = threadIdx.x >> 5, lane = threadIdx.x & 31;
  int y = blockIdx.y, xbase = blockIdx.x * 64;
  v8f acc[4] = {};
  for (int ks = 0; ks < 12; ++ks) {
    __syncthreads();
#pragma unroll
    for (int j = 0; j < 8; ++j) {
      int e = threadIdx.x * 8 + j;
      int k = e >> 6, x = e & 63;
      int c = ks * 32 + k;
      long s = (long)y * 256 + xbase + x;
      unsigned short v;
      if (c < 128)      v = po_b[s * 128 + c];
      else if (c < 256) v = pn_b[s * 128 + (c - 128)];
      else if (c < 320) v = featb[y * 64 + (c - 256)];           // left
      else              v = featb[(xbase + x) * 64 + (c - 320)]; // right
      Xf[x >> 4][((k >> 4) << 4) + (x & 15)][k & 15] = v;
    }
    __syncthreads();
    BF16Frag af;
    LOAD_FRAG(af, w1f + (((ks * 8) + w) * 32 + lane) * 16);
#pragma unroll
    for (int nt = 0; nt < 4; ++nt) {
      BF16Frag bf;
      LOAD_FRAG(bf, &Xf[nt][lane][0]);
      acc[nt] = __builtin_amdgcn_wmma_f32_16x16x32_bf16(
          false, af.v, false, bf.v, (short)0, acc[nt], false, false);
    }
  }
#pragma unroll
  for (int nt = 0; nt < 4; ++nt) {
    int col = nt * 16 + (lane & 15);
#pragma unroll
    for (int r = 0; r < 8; ++r) {
      int o = w * 16 + r + ((lane >> 4) << 3);
      x0[(long)o * 65536 + y * 256 + xbase + col] = acc[nt][r];
    }
  }
}

// ---------------------------------------------------------------------------
// 3x3 SAME conv as implicit WMMA GEMM.  Input slab stored [row][x][channel]
// so every tap fragment is two ds_load_b128 despite the dx shift.
// ---------------------------------------------------------------------------
__global__ __launch_bounds__(256) void k_conv3x3(
    const unsigned short* __restrict__ w3f,  // A-frag-major [36][8][32][16]
    const unsigned short* __restrict__ inb,  // [128][65536] bf16
    float* __restrict__ out) {               // [128][65536]
  __shared__ __align__(16) unsigned short Isl[3][68][32];  // 13 KB
  int w = threadIdx.x >> 5, lane = threadIdx.x & 31;
  int y = blockIdx.y, xbase = blockIdx.x * 64;
  v8f acc[4] = {};
  for (int cc = 0; cc < 4; ++cc) {
    __syncthreads();
    for (int e = threadIdx.x; e < 32 * 3 * 68; e += 256) {
      int k = e / 204, rem = e % 204;
      int r = rem / 68, cx = rem % 68;
      unsigned short v = 0;
      int gx = xbase - 1 + cx, gy = y + r - 1;
      if (cx < 66 && gx >= 0 && gx < 256 && gy >= 0 && gy < 256)
        v = inb[(long)(cc * 32 + k) * 65536 + gy * 256 + gx];
      Isl[r][cx][k] = v;
    }
    __syncthreads();
    if (cc < 3)  // gfx1250 global_prefetch of next channel chunk
      __builtin_prefetch(inb + (long)((cc + 1) * 32) * 65536 + (long)y * 256 + xbase, 0, 1);
#pragma unroll
    for (int dy = 0; dy < 3; ++dy) {
#pragma unroll
      for (int dx = 0; dx < 3; ++dx) {
        BF16Frag af;
        LOAD_FRAG(af, w3f + ((((cc * 9 + dy * 3 + dx) * 8) + w) * 32 + lane) * 16);
#pragma unroll
        for (int nt = 0; nt < 4; ++nt) {
          BF16Frag bf;
          LOAD_FRAG(bf, &Isl[dy][nt * 16 + (lane & 15) + dx][(lane >> 4) * 16]);
          acc[nt] = __builtin_amdgcn_wmma_f32_16x16x32_bf16(
              false, af.v, false, bf.v, (short)0, acc[nt], false, false);
        }
      }
    }
  }
#pragma unroll
  for (int nt = 0; nt < 4; ++nt) {
    int col = nt * 16 + (lane & 15);
#pragma unroll
    for (int r = 0; r < 8; ++r) {
      int o = w * 16 + r + ((lane >> 4) << 3);
      out[(long)o * 65536 + y * 256 + xbase + col] = acc[nt][r];
    }
  }
}

// ---------------------------------------------------------------------------
// InstanceNorm statistics: one block per channel.
// ---------------------------------------------------------------------------
__global__ __launch_bounds__(256) void k_inorm_stats(
    const float* __restrict__ src, float* __restrict__ stats) {
  __shared__ float ps[8], pq[8];
  int c = blockIdx.x;
  float s = 0.f, q = 0.f;
  for (int i = threadIdx.x; i < 65536; i += 256) {
    float v = src[(long)c * 65536 + i];
    s += v; q += v * v;
  }
  s = wave_sum(s); q = wave_sum(q);
  int w = threadIdx.x >> 5, lane = threadIdx.x & 31;
  if (lane == 0) { ps[w] = s; pq[w] = q; }
  __syncthreads();
  if (threadIdx.x == 0) {
    float ts = 0.f, tq = 0.f;
#pragma unroll
    for (int i = 0; i < 8; ++i) { ts += ps[i]; tq += pq[i]; }
    float mean = ts * (1.0f / 65536.0f);
    float var  = tq * (1.0f / 65536.0f) - mean * mean;
    stats[c * 2] = mean;
    stats[c * 2 + 1] = rsqrtf(var + IN_EPS);
  }
}

// IN + ELU pointwise; emits fp32 (optional) and bf16.
__global__ void k_in_elu(const float* __restrict__ src,
                         const float* __restrict__ stats,
                         const float* __restrict__ g, const float* __restrict__ b,
                         float* __restrict__ dstF, unsigned short* __restrict__ dstB) {
  int i = blockIdx.x * 256 + threadIdx.x;
  int c = i >> 16;
  float v = (src[i] - stats[c * 2]) * stats[c * 2 + 1] * g[c] + b[c];
  float e = eluf(v);
  if (dstF) dstF[i] = e;
  dstB[i] = f2bf(e);
}

// elu(x + IN(h2)) with NCHW->NHWC transpose into d_out
__global__ void k_final(const float* __restrict__ x1f, const float* __restrict__ h2,
                        const float* __restrict__ stats,
                        const float* __restrict__ g, const float* __restrict__ b,
                        float* __restrict__ out) {
  int i = blockIdx.x * 256 + threadIdx.x;
  int c = i >> 16, s = i & 65535;
  float v = x1f[i] + (h2[i] - stats[c * 2]) * stats[c * 2 + 1] * g[c] + b[c];
  out[(long)s * 128 + c] = eluf(v);
}

// ---------------------------------------------------------------------------
extern "C" void kernel_launch(void* const* d_in, const int* in_sizes, int n_in,
                              void* d_out, int out_size, void* d_ws, size_t ws_size,
                              hipStream_t stream) {
  const float* msa      = (const float*)d_in[0];
  const float* pair_o   = (const float*)d_in[1];
  const float* ln1_g    = (const float*)d_in[2];
  const float* ln1_b    = (const float*)d_in[3];
  const float* w1       = (const float*)d_in[4];
  const float* b1       = (const float*)d_in[5];
  const float* ln2d_g   = (const float*)d_in[6];
  const float* ln2d_b   = (const float*)d_in[7];
  const float* w2       = (const float*)d_in[8];
  const float* b2       = (const float*)d_in[9];
  const float* lnd_g    = (const float*)d_in[10];
  const float* lnd_b    = (const float*)d_in[11];
  const float* lno_g    = (const float*)d_in[12];
  const float* lno_b    = (const float*)d_in[13];
  const float* lnn_g    = (const float*)d_in[14];
  const float* lnn_b    = (const float*)d_in[15];
  const float* conv_in_w = (const float*)d_in[16];
  const float* in0_g    = (const float*)d_in[17];
  const float* in0_b    = (const float*)d_in[18];
  const float* rb_w1    = (const float*)d_in[19];
  const float* rb_in1_g = (const float*)d_in[20];
  const float* rb_in1_b = (const float*)d_in[21];
  const float* rb_w2    = (const float*)d_in[22];
  const float* rb_in2_g = (const float*)d_in[23];
  const float* rb_in2_b = (const float*)d_in[24];

  char* ws = (char*)d_ws;
  size_t off = 0;
  auto carve = [&](size_t bytes) -> void* {
    size_t a = (off + 255) & ~(size_t)255;
    off = a + bytes;
    return (void*)(ws + a);
  };
  unsigned short* xdb   = (unsigned short*)carve(192 * 8192 * 2);
  float*          xdf   = (float*)carve(192 * 256 * 32 * 4);
  unsigned short* featb = (unsigned short*)carve(256 * 64 * 2);
  unsigned short* w2f   = (unsigned short*)carve(32 * 8 * 32 * 16 * 2);
  unsigned short* w1f   = (unsigned short*)carve(12 * 8 * 32 * 16 * 2);
  unsigned short* w3f1  = (unsigned short*)carve(36 * 8 * 32 * 16 * 2);
  unsigned short* w3f2  = (unsigned short*)carve(36 * 8 * 32 * 16 * 2);
  unsigned short* po_b  = (unsigned short*)carve((size_t)65536 * 128 * 2);
  unsigned short* pn_b  = (unsigned short*)carve((size_t)65536 * 128 * 2);
  float*          x0    = (float*)carve((size_t)128 * 65536 * 4);
  float*          x1f   = (float*)carve((size_t)128 * 65536 * 4);
  unsigned short* x1b   = (unsigned short*)carve((size_t)128 * 65536 * 2);
  float*          hbuf  = (float*)carve((size_t)128 * 65536 * 4);
  unsigned short* h1b   = (unsigned short*)carve((size_t)128 * 65536 * 2);
  float*          st0   = (float*)carve(256 * 4);
  float*          st1   = (float*)carve(256 * 4);
  float*          st2   = (float*)carve(256 * 4);
  float* out = (float*)d_out;

  // weight preconversion into fragment-major bf16
  k_prep_w2frag<<<512, 256, 0, stream>>>(w2, w2f);
  k_prep_w1x1frag<<<192, 256, 0, stream>>>(conv_in_w, w1f);
  k_prep_w3x3frag<<<576, 256, 0, stream>>>(rb_w1, w3f1);
  k_prep_w3x3frag<<<576, 256, 0, stream>>>(rb_w2, w3f2);

  // MSA LN + proj, 1D features, pair_orig LN
  k_ln1_proj<<<192 * 256 / 8, 256, 0, stream>>>(msa, ln1_g, ln1_b, w1, b1, xdf, xdb);
  k_feat1d<<<256, 256, 0, stream>>>(xdf, lnd_g, lnd_b, featb);
  k_row_ln128_bf16<<<65536 / 8, 256, 0, stream>>>(pair_o, lno_g, lno_b, po_b);

  // fused outer-product + LN2d + proj2 + LN_new
  k_pair_fused<<<dim3(64, 64), 256, 0, stream>>>(xdb, ln2d_g, ln2d_b, w2f, b2,
                                                 lnn_g, lnn_b, pn_b);

  // ResidualNetwork
  k_conv1x1<<<dim3(4, 256), 256, 0, stream>>>(w1f, po_b, pn_b, featb, x0);
  k_inorm_stats<<<128, 256, 0, stream>>>(x0, st0);
  k_in_elu<<<(128 * 65536) / 256, 256, 0, stream>>>(x0, st0, in0_g, in0_b, x1f, x1b);

  k_conv3x3<<<dim3(4, 256), 256, 0, stream>>>(w3f1, x1b, hbuf);
  k_inorm_stats<<<128, 256, 0, stream>>>(hbuf, st1);
  k_in_elu<<<(128 * 65536) / 256, 256, 0, stream>>>(hbuf, st1, rb_in1_g, rb_in1_b,
                                                    (float*)nullptr, h1b);

  k_conv3x3<<<dim3(4, 256), 256, 0, stream>>>(w3f2, h1b, hbuf);
  k_inorm_stats<<<128, 256, 0, stream>>>(hbuf, st2);
  k_final<<<(128 * 65536) / 256, 256, 0, stream>>>(x1f, hbuf, st2, rb_in2_g, rb_in2_b, out);
}
